// BidirectionalRNN_72000831750678
// MI455X (gfx1250) — matrix operand order repaired and checked by reading, not compiled, not run on hardware
//
#include <hip/hip_runtime.h>
#include <hip/hip_bf16.h>
#include <math.h>

// Problem constants (match the JAX reference)
#define BATCH 64
#define TLEN  1024
#define DDIM  512
#define HDIM  512

typedef __attribute__((ext_vector_type(16))) __bf16 v16bf;
typedef __attribute__((ext_vector_type(8)))  float  v8f;

// ---------------------------------------------------------------------------
// Small prep kernels: f32 -> bf16 weight conversion, bias fusion, h0 = 0
// ---------------------------------------------------------------------------
__global__ void cvt_f32_bf16_kernel(const float* __restrict__ src,
                                    __bf16* __restrict__ dst, int n) {
    int i = blockIdx.x * blockDim.x + threadIdx.x;
    if (i < n) dst[i] = (__bf16)src[i];
}

__global__ void bias_sum_kernel(const float* __restrict__ a,
                                const float* __restrict__ b,
                                float* __restrict__ o, int n) {
    int i = blockIdx.x * blockDim.x + threadIdx.x;
    if (i < n) o[i] = a[i] + b[i];
}

__global__ void zero_bf16_kernel(__bf16* __restrict__ p, int n) {
    int i = blockIdx.x * blockDim.x + threadIdx.x;
    if (i < n) p[i] = (__bf16)0.0f;
}

// ---------------------------------------------------------------------------
// Phase 1: input projection  out[b,dir,t,:] = x_row . Wih_dir^T + (bih+bhh)
// One wave32 per 16x64 tile: one shared A fragment feeds 4 WMMAs per K-step,
// amortizing the f32->bf16 conversion of x 4x and cutting A traffic 4x.
// K=512 -> 4 x 16 = 64 v_wmma_f32_16x16x32_bf16 per wave.
// grid = (B*T/16, H/64, 2), block = 32
// ---------------------------------------------------------------------------
#define NT 4   // N-tiles (of 16) per wave

__global__ __launch_bounds__(32) void proj_wmma_kernel(
    const float*  __restrict__ x,          // [B,T,D] f32
    const __bf16* __restrict__ Wih_f_bf,   // [H,D] bf16
    const __bf16* __restrict__ Wih_b_bf,   // [H,D] bf16
    const float*  __restrict__ bias_f,     // [H] bih+bhh
    const float*  __restrict__ bias_b,     // [H]
    float*        __restrict__ out)        // [B,2,T,H]
{
    const int lane  = threadIdx.x;
    const int r     = lane & 15;     // row/col within half-wave
    const int khalf = lane >> 4;     // 0 or 1

    const int m0  = blockIdx.x * 16;        // row in (B*T) processing order
    const int n0  = blockIdx.y * (16 * NT); // column base in H
    const int dir = blockIdx.z;

    const int bidx = m0 / TLEN;          // 16-row tiles never straddle batches
    const int s0   = m0 % TLEN;          // processing-order timestep base

    const __bf16* W    = dir ? Wih_b_bf : Wih_f_bf;
    const float*  bias = dir ? bias_b  : bias_f;

    // A fragment source: lane (and lane+16) cover row s0 + (lane&15)
    const int s_row = s0 + r;
    const int x_row = bidx * TLEN + (dir ? (TLEN - 1 - s_row) : s_row);
    const float* xa = x + (size_t)x_row * DDIM;

    // B fragment sources: column n of output = row n of Wih (B = Wih^T)
    const __bf16* wp[NT];
#pragma unroll
    for (int j = 0; j < NT; ++j)
        wp[j] = W + (size_t)(n0 + j * 16 + r) * DDIM;

    v8f acc[NT] = {};
    for (int kk = 0; kk < DDIM; kk += 32) {
        v16bf a;
        // A 16x32 bf16 layout: lanes 0-15: K 0-7 & 16-23; lanes 16-31: K 8-15 & 24-31
#pragma unroll
        for (int i = 0; i < 8; ++i) {
            a[i]     = (__bf16)xa[kk + khalf * 8 + i];
            a[8 + i] = (__bf16)xa[kk + 16 + khalf * 8 + i];
        }
        // B 32x16 bf16: lane holds column n; lanes 0-15 K 0-7 & 16-23; lanes 16-31 K 8-15 & 24-31
#pragma unroll
        for (int j = 0; j < NT; ++j) {
            v16bf bm;
#pragma unroll
            for (int i = 0; i < 8; ++i) {
                bm[i]     = wp[j][kk + khalf * 8 + i];
                bm[8 + i] = wp[j][kk + 16 + khalf * 8 + i];
            }
            acc[j] = __builtin_amdgcn_wmma_f32_16x16x32_bf16(
                /*neg_a=*/false, a, /*neg_b=*/false, bm,
                /*c_mod=*/(short)0, acc[j], /*reuse_a=*/false, /*reuse_b=*/false);
        }
    }

    // C/D layout: VGPR rr -> M = rr + 8*khalf, N = lane&15
#pragma unroll
    for (int j = 0; j < NT; ++j) {
        const int ncol = n0 + j * 16 + r;
        const float bn = bias[ncol];
#pragma unroll
        for (int rr = 0; rr < 8; ++rr) {
            const int t = s0 + rr + 8 * khalf;
            const size_t oidx =
                (((size_t)bidx * 2 + dir) * TLEN + t) * (size_t)HDIM + ncol;
            out[oidx] = acc[j][rr] + bn;
        }
    }
}

// ---------------------------------------------------------------------------
// Phase 2: one recurrent step (both directions).
//   h_new = tanh(xp[:,t,:] + h_prev @ Whh^T), xp read from (and h written to)
//   d_out in place; h ping-pongs as bf16 in workspace keyed by t&1.
// grid = (B/16, H/16, 2), block = 32 (one wave per 16x16 tile) -- kept narrow
// to maximize wave-level parallelism on the latency-critical serial chain.
// ---------------------------------------------------------------------------
__global__ __launch_bounds__(32) void rnn_step_wmma_kernel(
    float*        __restrict__ out,        // [B,2,T,H] (xp in, h out at step t)
    const __bf16* __restrict__ Whh_f_bf,   // [H,H] bf16
    const __bf16* __restrict__ Whh_b_bf,   // [H,H] bf16
    __bf16*       __restrict__ hbuf,       // [2 dir][2 parity][B*H] bf16
    int t)
{
    const int lane  = threadIdx.x;
    const int r     = lane & 15;
    const int khalf = lane >> 4;

    const int m0  = blockIdx.x * 16;   // batch rows
    const int n0  = blockIdx.y * 16;   // hidden columns
    const int dir = blockIdx.z;
    const int parity = t & 1;

    const __bf16* W     = dir ? Whh_b_bf : Whh_f_bf;
    const __bf16* h_in  = hbuf + ((size_t)dir * 2 + parity)       * (BATCH * HDIM);
    __bf16*       h_out = hbuf + ((size_t)dir * 2 + (parity ^ 1)) * (BATCH * HDIM);

    const __bf16* ha = h_in + (size_t)(m0 + r) * HDIM;   // A row for this lane
    const int ncol   = n0 + r;
    const __bf16* wp = W + (size_t)ncol * HDIM;          // B = Whh^T column

    v8f acc = {};
    for (int kk = 0; kk < HDIM; kk += 32) {
        v16bf a, bm;
#pragma unroll
        for (int i = 0; i < 8; ++i) {
            a[i]     = ha[kk + khalf * 8 + i];
            a[8 + i] = ha[kk + 16 + khalf * 8 + i];
        }
#pragma unroll
        for (int i = 0; i < 8; ++i) {
            bm[i]     = wp[kk + khalf * 8 + i];
            bm[8 + i] = wp[kk + 16 + khalf * 8 + i];
        }

        acc = __builtin_amdgcn_wmma_f32_16x16x32_bf16(
            false, a, false, bm, (short)0, acc, false, false);
    }

#pragma unroll
    for (int rr = 0; rr < 8; ++rr) {
        const int bb = m0 + rr + 8 * khalf;
        const size_t oidx = (((size_t)bb * 2 + dir) * TLEN + t) * (size_t)HDIM + ncol;
        const float h = tanhf(acc[rr] + out[oidx]);
        out[oidx] = h;                                   // final f32 hidden
        h_out[(size_t)bb * HDIM + ncol] = (__bf16)h;     // bf16 state for t+1
    }
}

// ---------------------------------------------------------------------------
// Host-side launch
// ---------------------------------------------------------------------------
extern "C" void kernel_launch(void* const* d_in, const int* in_sizes, int n_in,
                              void* d_out, int out_size, void* d_ws, size_t ws_size,
                              hipStream_t stream) {
    (void)in_sizes; (void)n_in; (void)out_size; (void)ws_size;

    const float* x     = (const float*)d_in[0];
    const float* Wih_f = (const float*)d_in[1];
    const float* Whh_f = (const float*)d_in[2];
    const float* bih_f = (const float*)d_in[3];
    const float* bhh_f = (const float*)d_in[4];
    const float* Wih_b = (const float*)d_in[5];
    const float* Whh_b = (const float*)d_in[6];
    const float* bih_b = (const float*)d_in[7];
    const float* bhh_b = (const float*)d_in[8];
    float* out = (float*)d_out;

    // Workspace layout (bytes), 256-aligned sections
    char* ws = (char*)d_ws;
    const size_t W_ELEMS = (size_t)HDIM * DDIM;          // 262144
    __bf16* wihf_bf = (__bf16*)(ws + 0);
    __bf16* whhf_bf = (__bf16*)(ws + 1 * W_ELEMS * 2);
    __bf16* wihb_bf = (__bf16*)(ws + 2 * W_ELEMS * 2);
    __bf16* whhb_bf = (__bf16*)(ws + 3 * W_ELEMS * 2);
    float*  bias_f  = (float*)(ws + 4 * W_ELEMS * 2);
    float*  bias_b  = bias_f + HDIM;
    __bf16* hbuf    = (__bf16*)((char*)(bias_b + HDIM) + 256);
    const int HBUF_ELEMS = 2 * 2 * BATCH * HDIM;         // dir x parity x B*H

    // 1) Convert weights to bf16, fuse biases, zero hidden state (both parities)
    {
        const int n = (int)W_ELEMS, blk = 256, grd = (n + blk - 1) / blk;
        cvt_f32_bf16_kernel<<<grd, blk, 0, stream>>>(Wih_f, wihf_bf, n);
        cvt_f32_bf16_kernel<<<grd, blk, 0, stream>>>(Whh_f, whhf_bf, n);
        cvt_f32_bf16_kernel<<<grd, blk, 0, stream>>>(Wih_b, wihb_bf, n);
        cvt_f32_bf16_kernel<<<grd, blk, 0, stream>>>(Whh_b, whhb_bf, n);
        bias_sum_kernel<<<(HDIM + 255) / 256, 256, 0, stream>>>(bih_f, bhh_f, bias_f, HDIM);
        bias_sum_kernel<<<(HDIM + 255) / 256, 256, 0, stream>>>(bih_b, bhh_b, bias_b, HDIM);
        zero_bf16_kernel<<<(HBUF_ELEMS + 255) / 256, 256, 0, stream>>>(hbuf, HBUF_ELEMS);
    }

    // 2) Input projection for both directions, written into d_out in place
    {
        dim3 grid(BATCH * TLEN / 16, HDIM / (16 * NT), 2);
        proj_wmma_kernel<<<grid, 32, 0, stream>>>(x, wihf_bf, wihb_bf,
                                                  bias_f, bias_b, out);
    }

    // 3) Sequential recurrence: one launch per timestep (graph-captured),
    //    both directions per launch; stream order carries the dependency.
    {
        dim3 grid(BATCH / 16, HDIM / 16, 2);
        for (int t = 0; t < TLEN; ++t) {
            rnn_step_wmma_kernel<<<grid, 32, 0, stream>>>(out, whhf_bf, whhb_bf,
                                                          hbuf, t);
        }
    }
}
